// SelfAttention_RoPE_43267500540228
// MI455X (gfx1250) — matrix operand order repaired and checked by reading, not compiled, not run on hardware
//
#include <hip/hip_runtime.h>
#include <hip/hip_bf16.h>

// ---------------- CDNA5 WMMA types / helpers ----------------
typedef __attribute__((ext_vector_type(16))) _Float16 v16h;
typedef __attribute__((ext_vector_type(2)))  _Float16 v2h;
typedef __attribute__((ext_vector_type(8)))  float    v8f;

#define EMBED 1024
#define NHEAD 16
#define HDIM  64
#define BATCH 2
#define SEQ   2048
#define MROWS (BATCH * SEQ)   // 4096

__device__ __forceinline__ v8f wmma16(v16h a, v16h b, v8f c) {
  return __builtin_amdgcn_wmma_f32_16x16x32_f16(false, a, false, b, (short)0, c,
                                                false, false);
}

// Pack two f32 into one 32-bit LDS store of two f16.
__device__ __forceinline__ void store_pk(v2h* dst, float a, float b) {
  v2h t;
  t[0] = (_Float16)a;
  t[1] = (_Float16)b;
  *dst = t;
}

// A fragment: 16x32 f16 (M x K). Lane<16 holds row=lane, K={0..7,16..23};
// lane>=16 holds row=lane-16, K={8..15,24..31}; two packed halves per VGPR.
__device__ __forceinline__ v16h load_frag_a(const _Float16* base, int ld) {
  const int lane = threadIdx.x & 31;
  const int row  = lane & 15;
  const int kb   = (lane >> 4) << 3;           // 0 or 8
  v16h a;
#pragma unroll
  for (int i = 0; i < 8; ++i) {
    const int k = ((i & 4) << 2) + kb + ((i & 3) << 1);  // {0,16}+kb+2*(i%4)
    a[2 * i]     = base[row * ld + k];
    a[2 * i + 1] = base[row * ld + k + 1];
  }
  return a;
}

// B fragment: 32x16 f16 (K x N), source stored column-major: base[col*ld + k].
// Lane<16 holds col=lane, K={0..7,16..23}; lane>=16 holds col=lane-16, K={8..15,24..31}.
__device__ __forceinline__ v16h load_frag_b(const _Float16* base, int ld) {
  const int lane = threadIdx.x & 31;
  const int col  = lane & 15;
  const int kb   = (lane >> 4) << 3;           // 0 or 8
  v16h b;
#pragma unroll
  for (int i = 0; i < 8; ++i) {
    const int k = ((i & 4) << 2) + kb + ((i & 3) << 1);
    b[2 * i]     = base[col * ld + k];
    b[2 * i + 1] = base[col * ld + k + 1];
  }
  return b;
}

// Row-wise reductions over the 16-lane half-group that shares a C-matrix row.
__device__ __forceinline__ float rowmax16(float v) {
  v = fmaxf(v, __shfl_xor(v, 1, 32));
  v = fmaxf(v, __shfl_xor(v, 2, 32));
  v = fmaxf(v, __shfl_xor(v, 4, 32));
  v = fmaxf(v, __shfl_xor(v, 8, 32));
  return v;
}
__device__ __forceinline__ float rowsum16(float v) {
  v += __shfl_xor(v, 1, 32);
  v += __shfl_xor(v, 2, 32);
  v += __shfl_xor(v, 4, 32);
  v += __shfl_xor(v, 8, 32);
  return v;
}

// ---------------- Kernel 1: QKV GEMM + RoPE ----------------
// C[m, n] = x[m, :] @ w_qkv[:, n];  m in [0,4096), n in [0,3072).
// 256 threads (8 waves), tile 64(M) x 64(N), K-step 64 (4 WMMAs per barrier).
__global__ void __launch_bounds__(256)
qkv_rope_kernel(const float* __restrict__ x, const float* __restrict__ w_qkv,
                const float* __restrict__ freqs, _Float16* __restrict__ qws,
                _Float16* __restrict__ kws, _Float16* __restrict__ vws) {
  __shared__ __align__(16) _Float16 sA[64 * 64];   // [m][k]
  __shared__ __align__(16) _Float16 sB[64 * 64];   // [n][k] column-major

  const int n0  = blockIdx.x * 64;
  const int m0  = blockIdx.y * 64;
  const int tid = threadIdx.x;
  const int w   = tid >> 5;
  const int lane = tid & 31;
  const int m_off = (w & 3) * 16;
  const int n_off = (w >> 2) * 32;

  v8f c0 = {}, c1 = {};

  for (int k0 = 0; k0 < EMBED; k0 += 64) {
    __syncthreads();
#pragma unroll
    for (int i = 0; i < 8; ++i) {   // A tile 64x64: f32 pairs -> packed f16
      const int idx = tid + i * 256;           // pair index, 2048 pairs
      const int r = idx >> 5, c2 = idx & 31;   // 32 pairs per row
      const float2 xv =
          *(const float2*)&x[(size_t)(m0 + r) * EMBED + k0 + 2 * c2];
      store_pk((v2h*)sA + idx, xv.x, xv.y);
    }
#pragma unroll
    for (int i = 0; i < 8; ++i) {   // B tile 64(K)x64(N) -> [n][k] packed
      const int idx = tid + i * 256;           // 2048 pairs
      const int k2 = idx >> 6, nn = idx & 63;  // k pair, col n
      const float b0 = w_qkv[(size_t)(k0 + 2 * k2) * 3072 + n0 + nn];
      const float b1 = w_qkv[(size_t)(k0 + 2 * k2 + 1) * 3072 + n0 + nn];
      store_pk((v2h*)sB + nn * 32 + k2, b0, b1);
    }
    __syncthreads();

    const v16h a0 = load_frag_a(sA + m_off * 64, 64);
    const v16h a1 = load_frag_a(sA + m_off * 64 + 32, 64);
    c0 = wmma16(a0, load_frag_b(sB + n_off * 64, 64), c0);
    c0 = wmma16(a1, load_frag_b(sB + n_off * 64 + 32, 64), c0);
    c1 = wmma16(a0, load_frag_b(sB + (n_off + 16) * 64, 64), c1);
    c1 = wmma16(a1, load_frag_b(sB + (n_off + 16) * 64 + 32, 64), c1);
  }

  // Epilogue: part/head uniform per fragment; RoPE pair partner via lane^1.
  const int colL  = lane & 15;
  const int rhalf = (lane >> 4) << 3;
#pragma unroll
  for (int f = 0; f < 2; ++f) {
    const v8f cf   = f ? c1 : c0;
    const int ng   = n0 + n_off + f * 16 + colL;
    const int part = ng >> 10;               // 0=q 1=k 2=v (wave-uniform)
    const int wi   = ng & 1023;
    const int h    = wi >> 6;
    const int d    = wi & 63;
    if (part < 2) {
      _Float16* dst = part ? kws : qws;
      const float sgn = (d & 1) ? 1.0f : -1.0f;  // branchless rope sign
#pragma unroll
      for (int i = 0; i < 8; ++i) {
        const int mg = m0 + m_off + rhalf + i;
        const int bb = mg >> 11;             // / SEQ
        const int l  = mg & (SEQ - 1);
        float val = cf[i];
        const float other = __shfl_xor(val, 1, 32);
        const float2 fc = *(const float2*)&freqs[(size_t)l * 64 + (d >> 1) * 2];
        val = fmaf(val, fc.x, sgn * fc.y * other);
        dst[(((size_t)(bb * NHEAD + h)) * SEQ + l) * HDIM + d] = (_Float16)val;
      }
    } else {
#pragma unroll
      for (int i = 0; i < 8; ++i) {
        const int mg = m0 + m_off + rhalf + i;
        const int bb = mg >> 11;
        const int l  = mg & (SEQ - 1);
        vws[(((size_t)(bb * NHEAD + h)) * HDIM + d) * SEQ + l] =
            (_Float16)cf[i];
      }
    }
  }
}

// ---------------- Kernel 2: flash attention per (b, h, 64 queries) --------
// 4 waves x 16 query rows; key tiles of 64 (16 WMMAs per barrier pair);
// online softmax in f32.
__global__ void __launch_bounds__(128)
attn_kernel(const _Float16* __restrict__ qws, const _Float16* __restrict__ kws,
            const _Float16* __restrict__ vws, const float* __restrict__ bias,
            _Float16* __restrict__ aout) {
  __shared__ __align__(16) _Float16 sK[64 * 64];    // [key][d]
  __shared__ __align__(16) _Float16 sV[64 * 64];    // [d][key]
  __shared__ __align__(16) _Float16 sP[4][16 * 64]; // per-wave P [row][key]

  const int bb = blockIdx.z, h = blockIdx.y, q0 = blockIdx.x * 64;
  const int tid = threadIdx.x, w = tid >> 5, lane = tid & 31;
  const int colL = lane & 15, rhalf = (lane >> 4) << 3;
  const size_t kvbase = (size_t)(bb * NHEAD + h) * SEQ * HDIM;

  // Q fragments for this wave's 16 rows (d 0..31 and 32..63), loaded once.
  const _Float16* qbase = qws + kvbase + (size_t)(q0 + w * 16) * HDIM;
  const v16h aq0 = load_frag_a(qbase, HDIM);
  const v16h aq1 = load_frag_a(qbase + 32, HDIM);

  float mrun[8], lrun[8];
#pragma unroll
  for (int i = 0; i < 8; ++i) { mrun[i] = -3.0e38f; lrun[i] = 0.f; }
  v8f o0 = {}, o1 = {}, o2 = {}, o3 = {};
  const float scale = 0.125f;            // 1/sqrt(64)
  const float* bq = bias + ((size_t)h * SEQ + q0 + w * 16) * SEQ;

  for (int k0 = 0; k0 < SEQ; k0 += 64) {
    __syncthreads();
    // K tile: rows k0..k0+63 contiguous in global -> straight b128 copy.
    {
      const uint4* src = (const uint4*)(kws + kvbase + (size_t)k0 * HDIM);
      uint4* dst = (uint4*)sK;           // 512 uint4 total
#pragma unroll
      for (int i = 0; i < 4; ++i) dst[tid + i * 128] = src[tid + i * 128];
    }
    // V tile [d][key]: 64 rows x 64 halfs (8 uint4 per row).
    {
      uint4* dst = (uint4*)sV;
#pragma unroll
      for (int i = 0; i < 4; ++i) {
        const int idx = tid + i * 128;   // 512 uint4
        const int d = idx >> 3, q = idx & 7;
        dst[idx] =
            ((const uint4*)(vws + kvbase + (size_t)d * SEQ + k0))[q];
      }
    }
    __syncthreads();

    // Scores: S[16q x 64k] = Q(16x64) @ K^T, four 16x16 C frags.
    v8f s0 = {}, s1 = {}, s2 = {}, s3 = {};
    s0 = wmma16(aq0, load_frag_b(sK + 0 * 16 * 64, 64), s0);
    s0 = wmma16(aq1, load_frag_b(sK + 0 * 16 * 64 + 32, 64), s0);
    s1 = wmma16(aq0, load_frag_b(sK + 1 * 16 * 64, 64), s1);
    s1 = wmma16(aq1, load_frag_b(sK + 1 * 16 * 64 + 32, 64), s1);
    s2 = wmma16(aq0, load_frag_b(sK + 2 * 16 * 64, 64), s2);
    s2 = wmma16(aq1, load_frag_b(sK + 2 * 16 * 64 + 32, 64), s2);
    s3 = wmma16(aq0, load_frag_b(sK + 3 * 16 * 64, 64), s3);
    s3 = wmma16(aq1, load_frag_b(sK + 3 * 16 * 64 + 32, 64), s3);

    float p0[8], p1[8], p2[8], p3[8], corr[8];
#pragma unroll
    for (int i = 0; i < 8; ++i) {
      const int rm = rhalf + i;
      const float* br = bq + (size_t)rm * SEQ + k0 + colL;
      const float t0 = fmaf(s0[i], scale, br[0]);
      const float t1 = fmaf(s1[i], scale, br[16]);
      const float t2 = fmaf(s2[i], scale, br[32]);
      const float t3 = fmaf(s3[i], scale, br[48]);
      float mx = rowmax16(fmaxf(fmaxf(t0, t1), fmaxf(t2, t3)));
      mx = fmaxf(mx, mrun[i]);
      const float e0 = __expf(t0 - mx);
      const float e1 = __expf(t1 - mx);
      const float e2 = __expf(t2 - mx);
      const float e3 = __expf(t3 - mx);
      corr[i] = __expf(mrun[i] - mx);
      lrun[i] = lrun[i] * corr[i] + rowsum16((e0 + e1) + (e2 + e3));
      mrun[i] = mx;
      p0[i] = e0; p1[i] = e1; p2[i] = e2; p3[i] = e3;
    }
#pragma unroll
    for (int i = 0; i < 8; ++i) {        // rescale running O
      o0[i] *= corr[i]; o1[i] *= corr[i]; o2[i] *= corr[i]; o3[i] *= corr[i];
    }

    // Re-layout P (C frags) -> A frags through per-wave LDS.
    _Float16* pp = &sP[w][0];
#pragma unroll
    for (int i = 0; i < 8; ++i) {
      const int rm = rhalf + i;
      pp[rm * 64 + 0 + colL]  = (_Float16)p0[i];
      pp[rm * 64 + 16 + colL] = (_Float16)p1[i];
      pp[rm * 64 + 32 + colL] = (_Float16)p2[i];
      pp[rm * 64 + 48 + colL] = (_Float16)p3[i];
    }
    const v16h ap0 = load_frag_a(pp, 64);       // keys 0..31
    const v16h ap1 = load_frag_a(pp + 32, 64);  // keys 32..63
    o0 = wmma16(ap0, load_frag_b(sV + 0 * 16 * 64, 64), o0);
    o0 = wmma16(ap1, load_frag_b(sV + 0 * 16 * 64 + 32, 64), o0);
    o1 = wmma16(ap0, load_frag_b(sV + 1 * 16 * 64, 64), o1);
    o1 = wmma16(ap1, load_frag_b(sV + 1 * 16 * 64 + 32, 64), o1);
    o2 = wmma16(ap0, load_frag_b(sV + 2 * 16 * 64, 64), o2);
    o2 = wmma16(ap1, load_frag_b(sV + 2 * 16 * 64 + 32, 64), o2);
    o3 = wmma16(ap0, load_frag_b(sV + 3 * 16 * 64, 64), o3);
    o3 = wmma16(ap1, load_frag_b(sV + 3 * 16 * 64 + 32, 64), o3);
  }

  // Normalize and store attention output as f16 [b*SEQ + l][EMBED].
#pragma unroll
  for (int i = 0; i < 8; ++i) {
    const int rm = rhalf + i;
    const float inv = lrun[i] > 0.f ? 1.0f / lrun[i] : 0.f;
    const size_t rowg =
        (size_t)(bb * SEQ + q0 + w * 16 + rm) * EMBED + h * HDIM;
    aout[rowg + 0 + colL]  = (_Float16)(o0[i] * inv);
    aout[rowg + 16 + colL] = (_Float16)(o1[i] * inv);
    aout[rowg + 32 + colL] = (_Float16)(o2[i] * inv);
    aout[rowg + 48 + colL] = (_Float16)(o3[i] * inv);
  }
}

// ---------------- Kernel 3: output projection + bias ----------------
__global__ void __launch_bounds__(256)
proj_kernel(const _Float16* __restrict__ aout, const float* __restrict__ w_proj,
            const float* __restrict__ b_proj, float* __restrict__ out) {
  __shared__ __align__(16) _Float16 sA[64 * 64];
  __shared__ __align__(16) _Float16 sB[64 * 64];

  const int n0 = blockIdx.x * 64;
  const int m0 = blockIdx.y * 64;
  const int tid = threadIdx.x;
  const int w = tid >> 5, lane = tid & 31;
  const int m_off = (w & 3) * 16;
  const int n_off = (w >> 2) * 32;

  v8f c0 = {}, c1 = {};

  for (int k0 = 0; k0 < EMBED; k0 += 64) {
    __syncthreads();
#pragma unroll
    for (int i = 0; i < 2; ++i) {   // A tile 64x64 f16: b128 copies
      const int idx = tid + i * 256;           // 512 uint4
      const int r = idx >> 3, q = idx & 7;
      ((uint4*)sA)[idx] =
          ((const uint4*)(aout + (size_t)(m0 + r) * EMBED + k0))[q];
    }
#pragma unroll
    for (int i = 0; i < 8; ++i) {   // B tile 64(K)x64(N) -> [n][k] packed
      const int idx = tid + i * 256;
      const int k2 = idx >> 6, nn = idx & 63;
      const float b0 = w_proj[(size_t)(k0 + 2 * k2) * EMBED + n0 + nn];
      const float b1 = w_proj[(size_t)(k0 + 2 * k2 + 1) * EMBED + n0 + nn];
      store_pk((v2h*)sB + nn * 32 + k2, b0, b1);
    }
    __syncthreads();

    const v16h a0 = load_frag_a(sA + m_off * 64, 64);
    const v16h a1 = load_frag_a(sA + m_off * 64 + 32, 64);
    c0 = wmma16(a0, load_frag_b(sB + n_off * 64, 64), c0);
    c0 = wmma16(a1, load_frag_b(sB + n_off * 64 + 32, 64), c0);
    c1 = wmma16(a0, load_frag_b(sB + (n_off + 16) * 64, 64), c1);
    c1 = wmma16(a1, load_frag_b(sB + (n_off + 16) * 64 + 32, 64), c1);
  }

  const int colL = lane & 15, rhalf = (lane >> 4) << 3;
#pragma unroll
  for (int f = 0; f < 2; ++f) {
    const v8f cf = f ? c1 : c0;
    const int ng = n0 + n_off + f * 16 + colL;
    const float bv = b_proj[ng];
#pragma unroll
    for (int i = 0; i < 8; ++i) {
      const int mg = m0 + m_off + rhalf + i;
      out[(size_t)mg * EMBED + ng] = cf[i] + bv;
    }
  }
}

// ---------------- launch ----------------
extern "C" void kernel_launch(void* const* d_in, const int* in_sizes, int n_in,
                              void* d_out, int out_size, void* d_ws,
                              size_t ws_size, hipStream_t stream) {
  const float* x      = (const float*)d_in[0];
  const float* freqs  = (const float*)d_in[1];
  const float* bias   = (const float*)d_in[2];
  const float* w_qkv  = (const float*)d_in[3];
  const float* w_proj = (const float*)d_in[4];
  const float* b_proj = (const float*)d_in[5];
  float* out = (float*)d_out;

  const size_t per = (size_t)BATCH * NHEAD * SEQ * HDIM;  // 4,194,304 halves
  _Float16* qws  = (_Float16*)d_ws;
  _Float16* kws  = qws + per;
  _Float16* vws  = kws + per;
  _Float16* aout = vws + per;   // [4096][1024] f16

  qkv_rope_kernel<<<dim3(3072 / 64, MROWS / 64), 256, 0, stream>>>(
      x, w_qkv, freqs, qws, kws, vws);
  attn_kernel<<<dim3(SEQ / 64, NHEAD, BATCH), 128, 0, stream>>>(
      qws, kws, vws, bias, aout);
  proj_kernel<<<dim3(EMBED / 64, MROWS / 64), 256, 0, stream>>>(
      aout, w_proj, b_proj, out);
}
